// LocalEncoder_68685116997648
// MI455X (gfx1250) — matrix operand order repaired and checked by reading, not compiled, not run on hardware
//
#include <hip/hip_runtime.h>
#include <float.h>
#include <math.h>

typedef __attribute__((ext_vector_type(8)))  __bf16 v8bf;
typedef __attribute__((ext_vector_type(16))) __bf16 v16bf;
typedef __attribute__((ext_vector_type(8)))  float  v8f;

#define DIM    256
#define HEADS  8
#define DHEAD  32
#define WIN    128
#define NBLK   4
#define BATCH  8
#define SEQ    4096
#define NTOK   (BATCH * SEQ)      // 32768
#define NW     (SEQ / WIN)        // 32
#define FFI    682
#define FF1NR  1364               // 2*FFI (real)
#define FF1NP  1408               // padded to multiple of 128
#define FF2KP  704                // FFI padded to multiple of 32
#define ATT_SCALE 0.17677669529663687f   // 1/sqrt(32)

static __device__ __forceinline__ __bf16 to_bf(float f) { return (__bf16)f; }

static __device__ __forceinline__ v16bf ld_split16(const __bf16* p) {
    // two 16-byte chunks at p and p+16 elements -> one 16-element fragment
    v8bf lo = *(const v8bf*)p;
    v8bf hi = *(const v8bf*)(p + 16);
    return __builtin_shufflevector(lo, hi, 0,1,2,3,4,5,6,7,8,9,10,11,12,13,14,15);
}

// ---------------------------------------------------------------------------
// fp32 -> bf16 weight conversion with zero padding.
// dst: (NBLK, rd, cd) bf16 ; src: (NBLK, rs, cs) fp32
// ---------------------------------------------------------------------------
__global__ void cvt_pad(const float* __restrict__ src, __bf16* __restrict__ dst,
                        int rd, int cd, int rs, int cs) {
    size_t i = (size_t)blockIdx.x * 256 + threadIdx.x;
    size_t total = (size_t)NBLK * rd * cd;
    if (i >= total) return;
    int c = (int)(i % cd);
    int r = (int)((i / cd) % rd);
    int m = (int)(i / ((size_t)rd * cd));
    float v = (r < rs && c < cs) ? src[((size_t)m * rs + r) * cs + c] : 0.0f;
    dst[i] = to_bf(v);
}

// ---------------------------------------------------------------------------
// LayerNorm: one wave per row of 256, 8 rows per 256-thread block, bf16 out
// ---------------------------------------------------------------------------
__global__ __launch_bounds__(256)
void ln_bf16(const float* __restrict__ x, const float* __restrict__ g,
             const float* __restrict__ b, __bf16* __restrict__ out) {
    int row  = blockIdx.x * 8 + (threadIdx.x >> 5);
    int lane = threadIdx.x & 31;
    const float* xr = x + (size_t)row * DIM;
    float v[8], s = 0.f, s2 = 0.f;
#pragma unroll
    for (int j = 0; j < 8; ++j) {
        v[j] = xr[lane * 8 + j];
        s += v[j]; s2 += v[j] * v[j];
    }
#pragma unroll
    for (int d = 1; d < 32; d <<= 1) {
        s  += __shfl_xor(s,  d, 32);
        s2 += __shfl_xor(s2, d, 32);
    }
    float mu  = s  * (1.0f / DIM);
    float var = s2 * (1.0f / DIM) - mu * mu;
    float inv = rsqrtf(var + 1e-5f);
#pragma unroll
    for (int j = 0; j < 8; ++j) {
        int c = lane * 8 + j;
        out[(size_t)row * DIM + c] = to_bf((v[j] - mu) * inv * g[c] + b[c]);
    }
}

// ---------------------------------------------------------------------------
// Generic bf16 WMMA GEMM: C[M,N] = A[M,K] * W[N,K]^T
// 128 threads = 4 waves (2x2); each wave owns a 32x64 tile (2x4 WMMA 16x16x32)
// -> 8 WMMAs per 6 fragment loads per k-step (0.75 loads/WMMA).
// Block tile: 64 rows x 128 cols. N must be a multiple of 128, K of 32.
// mode 0: scatter QKV  -> qb (B,H,N,32), kb (B,H,N,32), vtb (B,H,32,N)
// mode 1: xbuf[m*DIM+e] += v   (residual accumulate, fp32)
// mode 2: obf[m*oNld+e] = bf16(v) for e < oNreal (FF1 hidden)
// ---------------------------------------------------------------------------
__global__ __launch_bounds__(128)
void gemm_bf16(const __bf16* __restrict__ A, int lda,
               const __bf16* __restrict__ W, int ldw,
               int N, int K, int mode,
               float* __restrict__ xbuf,
               __bf16* __restrict__ obf, int oNld, int oNreal,
               __bf16* __restrict__ qb, __bf16* __restrict__ kb,
               __bf16* __restrict__ vtb) {
    const int lane = threadIdx.x & 31;
    const int wv   = threadIdx.x >> 5;
    const int l16  = lane & 15;
    const int hi   = lane >> 4;             // which 16-lane half
    const int wm = wv >> 1, wn = wv & 1;    // 2x2 wave grid
    const int nTiles = N >> 7;              // 128-wide column blocks
    const int bm = (int)(blockIdx.x / nTiles) * 64;
    const int bn = (int)(blockIdx.x % nTiles) * 128;

    const int aK0 = hi ? 8 : 0;             // A-fragment K base per half-wave
    const int bK0 = hi ? 8 : 0;             // B-fragment K base per half-wave

    const int mrow[2] = { bm + wm * 32 + l16, bm + wm * 32 + 16 + l16 };
    const int ecol0   = bn + wn * 64;

    v8f acc[2][4] = {};
    for (int k0 = 0; k0 < K; k0 += 32) {
        v16bf af[2], wf[4];
#pragma unroll
        for (int t = 0; t < 2; ++t)
            af[t] = ld_split16(A + (size_t)mrow[t] * lda + k0 + aK0);
#pragma unroll
        for (int t = 0; t < 4; ++t) {
            int e = ecol0 + t * 16 + l16;
            wf[t] = ld_split16(W + (size_t)e * ldw + k0 + bK0);
        }
#pragma unroll
        for (int ti = 0; ti < 2; ++ti)
#pragma unroll
            for (int tj = 0; tj < 4; ++tj)
                acc[ti][tj] = __builtin_amdgcn_wmma_f32_16x16x32_bf16(
                    false, af[ti], false, wf[tj], (short)0, acc[ti][tj],
                    false, false);
    }

#pragma unroll
    for (int ti = 0; ti < 2; ++ti) {
#pragma unroll
        for (int tj = 0; tj < 4; ++tj) {
#pragma unroll
            for (int r = 0; r < 8; ++r) {
                int m = bm + wm * 32 + ti * 16 + r + hi * 8;
                int e = ecol0 + tj * 16 + l16;
                float v = acc[ti][tj][r];
                if (mode == 0) {
                    int s = e >> 8, rem = e & 255;
                    int hh = rem >> 5, d = rem & 31;
                    int b = m >> 12, n = m & (SEQ - 1);
                    size_t bh = (size_t)(b * HEADS + hh);
                    if (s == 0)      qb [(bh * SEQ + n) * DHEAD + d]  = to_bf(v);
                    else if (s == 1) kb [(bh * SEQ + n) * DHEAD + d]  = to_bf(v);
                    else             vtb[(bh * DHEAD + d) * SEQ + n]  = to_bf(v);
                } else if (mode == 1) {
                    xbuf[(size_t)m * DIM + e] += v;
                } else {
                    if (e < oNreal) obf[(size_t)m * oNld + e] = to_bf(v);
                }
            }
        }
    }
}

// ---------------------------------------------------------------------------
// Local windowed attention, fused: per wave = 16 query rows of one
// (batch, head, window). Scores 16x384 held in 24 register accumulators,
// softmax with half-wave shuffles, LDS transpose to A-layout, attn x V.
// Block = 128 threads (4 waves) covers 64 rows; 2 blocks per window.
// grid = B*H*NW*2 = 4096
// ---------------------------------------------------------------------------
__global__ __launch_bounds__(128)
void local_attn(const __bf16* __restrict__ qb, const __bf16* __restrict__ kb,
                const __bf16* __restrict__ vtb,
                const unsigned char* __restrict__ mask,
                __bf16* __restrict__ att) {
    __shared__ __align__(32) __bf16 attn_lds[4][16 * 384];

    const int lane = threadIdx.x & 31;
    const int wv   = threadIdx.x >> 5;
    const int l16  = lane & 15;
    const int hi   = lane >> 4;

    const int blk = blockIdx.x;
    const int rb = blk & 1;
    const int w  = (blk >> 1) & (NW - 1);
    const int h  = (blk >> 6) & (HEADS - 1);
    const int b  = blk >> 9;

    const int mbase = rb * 64 + wv * 16;        // query row offset in window
    const size_t bh = (size_t)(b * HEADS + h);
    const __bf16* qrow = qb + (bh * SEQ + (size_t)w * WIN) * DHEAD;
    const __bf16* krow = kb + bh * SEQ * DHEAD;
    const int jbase = (w - 1) * WIN;            // global key index of j=0

    // --- sim = q * k^T (K = DHEAD = 32 -> one WMMA per 16x16 tile) ---
    v16bf af = ld_split16(qrow + (size_t)(mbase + l16) * DHEAD + (hi ? 8 : 0));

    v8f acc[24];
#pragma unroll
    for (int t = 0; t < 24; ++t) {
        int idx = jbase + t * 16 + l16;
        v16bf kf = {};
        if (idx >= 0 && idx < SEQ)
            kf = ld_split16(krow + (size_t)idx * DHEAD + (hi ? 8 : 0));
        v8f z = {};
        acc[t] = __builtin_amdgcn_wmma_f32_16x16x32_bf16(
            false, af, false, kf, (short)0, z, false, false);
    }

    // --- scale + key mask ---
    const unsigned char* mrow = mask + (size_t)b * SEQ;
#pragma unroll
    for (int t = 0; t < 24; ++t) {
        int idx = jbase + t * 16 + l16;
        bool valid = (idx >= 0 && idx < SEQ);
        if (valid) valid = (mrow[idx] != 0);
#pragma unroll
        for (int r = 0; r < 8; ++r)
            acc[t][r] = valid ? acc[t][r] * ATT_SCALE : -FLT_MAX;
    }

    // --- softmax: lanes 0-15 of VGPR r hold row r, lanes 16-31 row r+8 ---
    float rsum[8];
#pragma unroll
    for (int r = 0; r < 8; ++r) {
        float mx = -FLT_MAX;
#pragma unroll
        for (int t = 0; t < 24; ++t) mx = fmaxf(mx, acc[t][r]);
#pragma unroll
        for (int d = 1; d < 16; d <<= 1) mx = fmaxf(mx, __shfl_xor(mx, d, 32));
        float s = 0.f;
#pragma unroll
        for (int t = 0; t < 24; ++t) {
            float p = (acc[t][r] > -FLT_MAX) ? expf(acc[t][r] - mx) : 0.f;
            acc[t][r] = p; s += p;
        }
#pragma unroll
        for (int d = 1; d < 16; d <<= 1) s += __shfl_xor(s, d, 32);
        rsum[r] = s;
    }
#pragma unroll
    for (int r = 0; r < 8; ++r) {
        float inv = rsum[r] > 0.f ? 1.0f / rsum[r] : 0.f;
        int row = r + hi * 8;
#pragma unroll
        for (int t = 0; t < 24; ++t)
            attn_lds[wv][row * 384 + t * 16 + l16] = to_bf(acc[t][r] * inv);
    }
    __syncthreads();

    // --- out = attn (16x384) * vv (384x32) ---
    const __bf16* vt = vtb + bh * DHEAD * SEQ;
    v8f acc2[2] = {};
#pragma unroll
    for (int ks = 0; ks < 12; ++ks) {
        v16bf a2 = ld_split16(&attn_lds[wv][l16 * 384 + ks * 32 + (hi ? 8 : 0)]);
        int jg = jbase + ks * 32 + (hi ? 8 : 0);    // 8-aligned split chunks
#pragma unroll
        for (int t2 = 0; t2 < 2; ++t2) {
            int d = t2 * 16 + l16;
            v16bf b2 = {};
            if (jg >= 0 && jg < SEQ)
                b2 = ld_split16(vt + (size_t)d * SEQ + jg);
            acc2[t2] = __builtin_amdgcn_wmma_f32_16x16x32_bf16(
                false, a2, false, b2, (short)0, acc2[t2], false, false);
        }
    }
#pragma unroll
    for (int t2 = 0; t2 < 2; ++t2)
#pragma unroll
        for (int r = 0; r < 8; ++r) {
            int n = w * WIN + mbase + r + hi * 8;
            int e = h * DHEAD + t2 * 16 + l16;
            att[((size_t)b * SEQ + n) * DIM + e] = to_bf(acc2[t2][r]);
        }
}

// ---------------------------------------------------------------------------
// GEGLU with exact GELU: y[:, :682] = a * 0.5*g*(1+erf(g/sqrt2)); pad to 704
// ---------------------------------------------------------------------------
__global__ void geglu(const __bf16* __restrict__ hbuf, __bf16* __restrict__ ybuf) {
    size_t i = (size_t)blockIdx.x * 256 + threadIdx.x;
    size_t total = (size_t)NTOK * FF2KP;
    if (i >= total) return;
    int c = (int)(i % FF2KP);
    int m = (int)(i / FF2KP);
    float y = 0.f;
    if (c < FFI) {
        float a = (float)hbuf[(size_t)m * FF1NP + c];
        float g = (float)hbuf[(size_t)m * FF1NP + c + FFI];
        y = a * 0.5f * g * (1.0f + erff(g * 0.70710678118654752f));
    }
    ybuf[i] = to_bf(y);
}

// ---------------------------------------------------------------------------
extern "C" void kernel_launch(void* const* d_in, const int* in_sizes, int n_in,
                              void* d_out, int out_size, void* d_ws, size_t ws_size,
                              hipStream_t stream) {
    const float* x            = (const float*)d_in[0];
    const unsigned char* mask = (const unsigned char*)d_in[1];
    const float* ln1g = (const float*)d_in[2];
    const float* ln1b = (const float*)d_in[3];
    const float* qkvw = (const float*)d_in[4];
    const float* outw = (const float*)d_in[5];
    const float* ln2g = (const float*)d_in[6];
    const float* ln2b = (const float*)d_in[7];
    const float* ffw1 = (const float*)d_in[8];
    const float* ffw2 = (const float*)d_in[9];

    char* ws = (char*)d_ws;
    float* xbuf = (float*)ws;            ws += (size_t)NTOK * DIM * 4;     // 33.6MB
    __bf16* xn  = (__bf16*)ws;           ws += (size_t)NTOK * DIM * 2;     // 16.8MB
    __bf16* qb  = (__bf16*)ws;           ws += (size_t)NTOK * DIM * 2;     // 16.8MB
    __bf16* kb  = (__bf16*)ws;           ws += (size_t)NTOK * DIM * 2;     // 16.8MB
    __bf16* vtb = (__bf16*)ws;           ws += (size_t)NTOK * DIM * 2;     // 16.8MB
    __bf16* attb= (__bf16*)ws;           ws += (size_t)NTOK * DIM * 2;     // 16.8MB
    __bf16* hbuf= (__bf16*)ws;           ws += (size_t)NTOK * FF1NP * 2;   // 92.3MB
    __bf16* wq  = (__bf16*)ws;           ws += (size_t)NBLK * 768 * 256 * 2;
    __bf16* wo  = (__bf16*)ws;           ws += (size_t)NBLK * 256 * 256 * 2;
    __bf16* w1  = (__bf16*)ws;           ws += (size_t)NBLK * FF1NP * 256 * 2;
    __bf16* w2  = (__bf16*)ws;           ws += (size_t)NBLK * 256 * FF2KP * 2;
    __bf16* ybuf = qb;  // alias: q/k/v are dead once FF stage runs (46MB < 50MB)

    hipMemcpyAsync(xbuf, x, (size_t)NTOK * DIM * 4, hipMemcpyDeviceToDevice, stream);

    // weight conversion (padded, zero-filled)
    {
        auto g = [](size_t n) { return (unsigned)((n + 255) / 256); };
        cvt_pad<<<g((size_t)NBLK*768*256),   256, 0, stream>>>(qkvw, wq, 768,  256, 768,  256);
        cvt_pad<<<g((size_t)NBLK*256*256),   256, 0, stream>>>(outw, wo, 256,  256, 256,  256);
        cvt_pad<<<g((size_t)NBLK*FF1NP*256), 256, 0, stream>>>(ffw1, w1, FF1NP,256, FF1NR,256);
        cvt_pad<<<g((size_t)NBLK*256*FF2KP), 256, 0, stream>>>(ffw2, w2, 256,  FF2KP,256, FFI);
    }

    const int MB = NTOK / 64; // 512 row-blocks of 64
    for (int i = 0; i < NBLK; ++i) {
        // --- attention sublayer ---
        ln_bf16<<<NTOK/8, 256, 0, stream>>>(xbuf, ln1g + i*DIM, ln1b + i*DIM, xn);
        gemm_bf16<<<MB * (768/128), 128, 0, stream>>>(
            xn, DIM, wq + (size_t)i*768*256, 256, 768, 256, /*mode*/0,
            nullptr, nullptr, 0, 0, qb, kb, vtb);
        local_attn<<<BATCH*HEADS*NW*2, 128, 0, stream>>>(qb, kb, vtb, mask, attb);
        gemm_bf16<<<MB * (256/128), 128, 0, stream>>>(
            attb, DIM, wo + (size_t)i*256*256, 256, 256, 256, /*mode*/1,
            xbuf, nullptr, 0, 0, nullptr, nullptr, nullptr);

        // --- GEGLU FF sublayer ---
        ln_bf16<<<NTOK/8, 256, 0, stream>>>(xbuf, ln2g + i*DIM, ln2b + i*DIM, xn);
        gemm_bf16<<<MB * (FF1NP/128), 128, 0, stream>>>(
            xn, DIM, w1 + (size_t)i*FF1NP*256, 256, FF1NP, 256, /*mode*/2,
            nullptr, hbuf, FF1NP, FF1NR, nullptr, nullptr, nullptr);
        geglu<<<(unsigned)(((size_t)NTOK*FF2KP + 255)/256), 256, 0, stream>>>(hbuf, ybuf);
        gemm_bf16<<<MB * (256/128), 128, 0, stream>>>(
            ybuf, FF2KP, w2 + (size_t)i*256*FF2KP, FF2KP, 256, FF2KP, /*mode*/1,
            xbuf, nullptr, 0, 0, nullptr, nullptr, nullptr);
    }

    hipMemcpyAsync(d_out, xbuf, (size_t)NTOK * DIM * 4, hipMemcpyDeviceToDevice, stream);
}
